// Stage_88235808129100
// MI455X (gfx1250) — compile-verified, matrix operand order checked by reading, not run on hardware
//
#include <hip/hip_runtime.h>

#define BB 4
#define NN 8192
#define CC 64
#define SS 2048
#define KK 32
#define C2 128
#define EXPAND 4
#define KP_SA 96     // pad of 67
#define KP_LA 160    // pad of 131

typedef __attribute__((ext_vector_type(16))) __bf16 bfx16;
typedef __attribute__((ext_vector_type(8)))  float  v8f;

union BFrag { uint4 q[2]; bfx16 v; };

__device__ __forceinline__ unsigned short f2bf(float x) {
    unsigned u = __builtin_bit_cast(unsigned, x);
    u += 0x7fffu + ((u >> 16) & 1u);           // round-to-nearest-even
    return (unsigned short)(u >> 16);
}

__device__ __forceinline__ bfx16 ld_frag(const unsigned short* p) {
    BFrag f;
    f.q[0] = *reinterpret_cast<const uint4*>(p);        // K elements [0..7]
    f.q[1] = *reinterpret_cast<const uint4*>(p + 16);   // K elements [16..23]
    return f.v;
}

// async global->LDS 16B copy (ASYNCcnt); LDS offset = low 32 bits of generic addr
__device__ __forceinline__ void async_copy_b128(unsigned short* ldsDst,
                                                const unsigned short* gbase,
                                                unsigned byteOff) {
    unsigned lds = (unsigned)(size_t)ldsDst;
    asm volatile("global_load_async_to_lds_b128 %0, %1, %2"
                 :: "v"(lds), "v"(byteOff),
                    "s"((unsigned long long)(size_t)gbase)
                 : "memory");
}
__device__ __forceinline__ void wait_async0() {
    asm volatile("s_wait_asynccnt 0" ::: "memory");
}

// ---------------- one-time weight convert + pad to bf16 ----------------
__global__ void k_prep_weights(const float* __restrict__ sa_w, const float* __restrict__ la_w,
                               const float* __restrict__ p1,   const float* __restrict__ p2,
                               unsigned short* __restrict__ wsa, unsigned short* __restrict__ wla,
                               unsigned short* __restrict__ w1,  unsigned short* __restrict__ w2) {
    int t = blockIdx.x * blockDim.x + threadIdx.x;
    const int NSA = C2 * KP_SA, NLA = C2 * KP_LA, N1 = (C2 * EXPAND) * C2, N2 = C2 * (C2 * EXPAND);
    if (t < NSA) {
        int n = t / KP_SA, k = t % KP_SA;
        wsa[t] = f2bf(k < (CC + 3) ? sa_w[n * (CC + 3) + k] : 0.f);
    } else if ((t -= NSA) < NLA) {
        int n = t / KP_LA, k = t % KP_LA;
        wla[t] = f2bf(k < (C2 + 3) ? la_w[n * (C2 + 3) + k] : 0.f);
    } else if ((t -= NLA) < N1) {
        w1[t] = f2bf(p1[t]);                   // 512x128, contiguous, no pad
    } else if ((t -= N1) < N2) {
        w2[t] = f2bf(p2[t]);                   // 128x512, contiguous, no pad
    }
}

// ---------------- input transpose: (B,3,N)->(B,N,3), (B,C,N)->(B,N,C) ----------------
__global__ void k_transpose_in(const float* __restrict__ pc, const float* __restrict__ pf,
                               float* __restrict__ coor, float* __restrict__ fea) {
    int t = blockIdx.x * blockDim.x + threadIdx.x;
    if (t >= BB * NN) return;
    int b = t / NN, n = t % NN;
#pragma unroll
    for (int d = 0; d < 3; ++d) coor[t * 3 + d] = pc[((size_t)b * 3 + d) * NN + n];
    for (int c = 0; c < CC; ++c) fea[(size_t)t * CC + c] = pf[((size_t)b * CC + c) * NN + n];
}

// ---------------- farthest point sampling: one block per batch ----------------
__global__ void k_fps(const float* __restrict__ coor, int* __restrict__ fpsidx) {
    __shared__ float dist[NN];      // 32 KB
    __shared__ float rv[1024];
    __shared__ int   ri[1024];
    __shared__ int   sfar;
    const int b = blockIdx.x, tid = threadIdx.x;
    for (int i = tid; i < NN; i += 1024) dist[i] = 1e10f;
    int far = 0;
    __syncthreads();
    for (int s = 0; s < SS; ++s) {
        if (tid == 0) fpsidx[b * SS + s] = far;
        const float cx = coor[((size_t)b * NN + far) * 3 + 0];
        const float cy = coor[((size_t)b * NN + far) * 3 + 1];
        const float cz = coor[((size_t)b * NN + far) * 3 + 2];
        float bd = -1.f; int bi = 0;
        for (int i = tid; i < NN; i += 1024) {
            float dx = coor[((size_t)b * NN + i) * 3 + 0] - cx;
            float dy = coor[((size_t)b * NN + i) * 3 + 1] - cy;
            float dz = coor[((size_t)b * NN + i) * 3 + 2] - cz;
            float d  = dx * dx + dy * dy + dz * dz;
            float nd = fminf(dist[i], d);
            dist[i] = nd;
            if (nd > bd) { bd = nd; bi = i; }
        }
        rv[tid] = bd; ri[tid] = bi;
        __syncthreads();
        for (int off = 512; off > 0; off >>= 1) {
            if (tid < off) {
                if (rv[tid + off] > rv[tid] ||
                    (rv[tid + off] == rv[tid] && ri[tid + off] < ri[tid])) {
                    rv[tid] = rv[tid + off]; ri[tid] = ri[tid + off];
                }
            }
            __syncthreads();
        }
        if (tid == 0) sfar = ri[0];
        __syncthreads();
        far = sfar;
    }
}

// ---------------- gather FPS centers ----------------
__global__ void k_newcoor(const float* __restrict__ coor, const int* __restrict__ fpsidx,
                          float* __restrict__ newc) {
    int t = blockIdx.x * blockDim.x + threadIdx.x;
    if (t >= BB * SS) return;
    int b = t / SS;
    int j = fpsidx[t];
#pragma unroll
    for (int d = 0; d < 3; ++d) newc[t * 3 + d] = coor[((size_t)b * NN + j) * 3 + d];
}

// ---------------- KNN top-32 with ball fallback ----------------
__global__ void k_knn(const float* __restrict__ pts, const float* __restrict__ qry,
                      int NP, float r2, int* __restrict__ outi) {
    int t = blockIdx.x * blockDim.x + threadIdx.x;   // B*S queries
    if (t >= BB * SS) return;
    int b = t / SS;
    const float qx = qry[t * 3 + 0], qy = qry[t * 3 + 1], qz = qry[t * 3 + 2];
    float bd[KK]; int bi[KK];
#pragma unroll
    for (int k = 0; k < KK; ++k) { bd[k] = 3.4e38f; bi[k] = 0; }
    const float* p = pts + (size_t)b * NP * 3;
    for (int j = 0; j < NP; ++j) {
        __builtin_prefetch(p + (j + 64) * 3, 0, 0);  // global_prefetch_b8
        float dx = p[j * 3 + 0] - qx;
        float dy = p[j * 3 + 1] - qy;
        float dz = p[j * 3 + 2] - qz;
        float d  = dx * dx + dy * dy + dz * dz;
        if (d < bd[KK - 1]) {
            int pos = KK - 1;
            while (pos > 0 && bd[pos - 1] > d) {
                bd[pos] = bd[pos - 1]; bi[pos] = bi[pos - 1]; --pos;
            }
            bd[pos] = d; bi[pos] = j;
        }
    }
    int nearest = bi[0];
#pragma unroll
    for (int k = 0; k < KK; ++k)
        outi[(size_t)t * KK + k] = (bd[k] > r2) ? nearest : bi[k];
}

// ---------------- fused gather + WMMA GEMM + bias + ReLU + max over K ----------------
// rows = (b,s,k) flat; A row = [fea[gidx] (CF), (pts[gidx]-ctr)/R (3), zero-pad]
// weights arrive pre-converted/padded bf16 -> staged via async global->LDS
template <int CF>
__global__ void k_gather_gemm(const float* __restrict__ fea, const float* __restrict__ pts,
                              const float* __restrict__ ctr, const int* __restrict__ gidx,
                              const unsigned short* __restrict__ wbf,
                              const float* __restrict__ bias,
                              int NP, float invR, float* __restrict__ outfea) {
    constexpr int DIN  = CF + 3;
    constexpr int KPAD = (DIN + 31) & ~31;
    constexpr int NKC  = KPAD / 32;
    constexpr int MT   = 64;                    // 4 waves x 16-row strips
    __shared__ alignas(16) unsigned short As[MT * KPAD];
    __shared__ alignas(16) unsigned short Wt[C2 * KPAD];
    __shared__ float smax[(MT / 16) * C2];
    const int tid  = threadIdx.x;
    const int row0 = blockIdx.x * MT;

    // async-stage bf16 weights (contiguous C2*KPAD elements)
    constexpr int NCHUNK = C2 * KPAD / 8;       // 16B chunks
    for (int ch = tid; ch < NCHUNK; ch += blockDim.x)
        async_copy_b128(&Wt[ch * 8], wbf, (unsigned)(ch * 16));

    // gather + convert A rows while the weight DMA is in flight
    for (int idx = tid; idx < MT * KPAD; idx += blockDim.x) {
        int r = idx / KPAD, c = idx % KPAD;
        int row = row0 + r;
        int b   = row / (SS * KK);
        int bs  = row / KK;                     // b*S + s
        int j   = gidx[row];
        float v = 0.f;
        if (c < CF)       v = fea[((size_t)b * NP + j) * CF + c];
        else if (c < DIN) v = (pts[((size_t)b * NP + j) * 3 + (c - CF)] -
                               ctr[(size_t)bs * 3 + (c - CF)]) * invR;
        As[idx] = f2bf(v);
    }
    wait_async0();
    __syncthreads();

    const int lane = tid & 31, wv = tid >> 5;
    const int half = lane >> 4, l16 = lane & 15;
    v8f acc[8] = {};
#pragma unroll
    for (int kc = 0; kc < NKC; ++kc) {
        const int kb = kc * 32 + half * 8;
        bfx16 a = ld_frag(&As[(wv * 16 + l16) * KPAD + kb]);
#pragma unroll
        for (int t = 0; t < 8; ++t) {
            bfx16 bm = ld_frag(&Wt[(t * 16 + l16) * KPAD + kb]);
            acc[t] = __builtin_amdgcn_wmma_f32_16x16x32_bf16(
                false, a, false, bm, (short)0, acc[t], false, false);
        }
    }
    // max over rows of relu(acc+bias): acc rows of this lane all share one s-group
#pragma unroll
    for (int t = 0; t < 8; ++t) {
        const int col = t * 16 + l16;
        const float bb = bias[col];
        float mx = 0.f;                         // ReLU floor == identity of max
#pragma unroll
        for (int r = 0; r < 8; ++r) mx = fmaxf(mx, acc[t][r] + bb);
        mx = fmaxf(mx, __shfl_xor(mx, 16));     // combine lane halves (16 rows)
        smax[wv * C2 + col] = mx;
    }
    __syncthreads();
    for (int idx = tid; idx < 2 * C2; idx += blockDim.x) {
        int g = idx >> 7, col = idx & (C2 - 1);
        float mv = fmaxf(smax[(2 * g) * C2 + col], smax[(2 * g + 1) * C2 + col]);
        int bs = (row0 + g * 32) / KK;
        outfea[(size_t)bs * C2 + col] = mv;
    }
}

// ---------------- dense WMMA GEMM: out = [relu](A * w^T + bias [+ res]) ----------------
// weights pre-converted bf16 (DOUT x DIN, contiguous) -> async-staged per K slice
template <bool RELU, bool RES, bool TROUT>
__global__ void k_gemm(const float* __restrict__ A, const unsigned short* __restrict__ wbf,
                       const float* __restrict__ bias, const float* __restrict__ res,
                       float* __restrict__ out, int DIN, int DOUT) {
    constexpr int MT = 64;
    __shared__ alignas(16) unsigned short As[MT * 128];
    __shared__ alignas(16) unsigned short Ws[128 * 128];
    const int tid  = threadIdx.x;
    const int row0 = blockIdx.x * MT;
    const int col0 = blockIdx.y * 128;
    const int lane = tid & 31, wv = tid >> 5;
    const int half = lane >> 4, l16 = lane & 15;
    v8f acc[8] = {};
    for (int ks = 0; ks < DIN; ks += 128) {
        // async DMA the 128x128 bf16 weight slice: row (col0+n), cols [ks,ks+128)
        for (int ch = tid; ch < 128 * 16; ch += blockDim.x) {
            int n = ch >> 4, c16 = ch & 15;
            async_copy_b128(&Ws[n * 128 + c16 * 8], wbf,
                            (unsigned)((((col0 + n) * DIN + ks) * 2) + c16 * 16));
        }
        // stage + convert A slice while DMA is in flight
        for (int idx = tid; idx < MT * 128; idx += blockDim.x) {
            int r = idx >> 7, c = idx & 127;
            As[idx] = f2bf(A[(size_t)(row0 + r) * DIN + ks + c]);
        }
        wait_async0();
        __syncthreads();
#pragma unroll
        for (int kc = 0; kc < 4; ++kc) {
            const int kb = kc * 32 + half * 8;
            bfx16 a = ld_frag(&As[(wv * 16 + l16) * 128 + kb]);
#pragma unroll
            for (int t = 0; t < 8; ++t) {
                bfx16 bm = ld_frag(&Ws[(t * 16 + l16) * 128 + kb]);
                acc[t] = __builtin_amdgcn_wmma_f32_16x16x32_bf16(
                    false, a, false, bm, (short)0, acc[t], false, false);
            }
        }
        __syncthreads();
    }
#pragma unroll
    for (int t = 0; t < 8; ++t) {
        const int col = col0 + t * 16 + l16;
        const float bb = bias[col];
#pragma unroll
        for (int r = 0; r < 8; ++r) {
            int rowG = row0 + wv * 16 + half * 8 + r;
            float v = acc[t][r] + bb;
            if (RES)  v += res[(size_t)rowG * DOUT + col];
            if (RELU) v = fmaxf(v, 0.f);
            if (TROUT) {
                int b = rowG / SS, s = rowG % SS;
                out[((size_t)b * DOUT + col) * SS + s] = v;   // (B,DOUT,S)
            } else {
                out[(size_t)rowG * DOUT + col] = v;
            }
        }
    }
}

// ---------------- transposed coord output (B,3,S) ----------------
__global__ void k_coor_out(const float* __restrict__ newc, float* __restrict__ out) {
    int t = blockIdx.x * blockDim.x + threadIdx.x;
    if (t >= BB * SS) return;
    int b = t / SS, s = t % SS;
#pragma unroll
    for (int d = 0; d < 3; ++d) out[((size_t)b * 3 + d) * SS + s] = newc[t * 3 + d];
}

extern "C" void kernel_launch(void* const* d_in, const int* in_sizes, int n_in,
                              void* d_out, int out_size, void* d_ws, size_t ws_size,
                              hipStream_t stream) {
    (void)in_sizes; (void)n_in; (void)out_size; (void)ws_size;
    const float* pc    = (const float*)d_in[0];
    const float* pf    = (const float*)d_in[1];
    const float* sa_w  = (const float*)d_in[2];
    const float* sa_b  = (const float*)d_in[3];
    const float* la_w  = (const float*)d_in[4];
    const float* la_b  = (const float*)d_in[5];
    const float* pw1_w = (const float*)d_in[6];
    const float* pw1_b = (const float*)d_in[7];
    const float* pw2_w = (const float*)d_in[8];
    const float* pw2_b = (const float*)d_in[9];
    float* ws = (float*)d_ws;
    size_t off = 0;
    float* coor   = ws + off; off += (size_t)BB * NN * 3;
    float* fea    = ws + off; off += (size_t)BB * NN * CC;
    int*   fpsidx = (int*)(ws + off); off += (size_t)BB * SS;
    float* newc   = ws + off; off += (size_t)BB * SS * 3;
    int*   gidx1  = (int*)(ws + off); off += (size_t)BB * SS * KK;
    float* nfea   = ws + off; off += (size_t)BB * SS * C2;
    int*   gidx2  = (int*)(ws + off); off += (size_t)BB * SS * KK;
    float* xla    = ws + off; off += (size_t)BB * SS * C2;
    float* h      = ws + off; off += (size_t)BB * SS * C2 * EXPAND;
    // bf16 weight buffers (element counts are multiples of 8 -> 16B alignment kept)
    unsigned short* wsa_bf = (unsigned short*)(ws + off); off += (size_t)C2 * KP_SA / 2;
    unsigned short* wla_bf = (unsigned short*)(ws + off); off += (size_t)C2 * KP_LA / 2;
    unsigned short* w1_bf  = (unsigned short*)(ws + off); off += (size_t)(C2 * EXPAND) * C2 / 2;
    unsigned short* w2_bf  = (unsigned short*)(ws + off); off += (size_t)C2 * (C2 * EXPAND) / 2;
    float* out    = (float*)d_out;

    const int NWTOT = C2 * KP_SA + C2 * KP_LA + (C2 * EXPAND) * C2 + C2 * (C2 * EXPAND);
    k_prep_weights<<<(NWTOT + 255) / 256, 256, 0, stream>>>(sa_w, la_w, pw1_w, pw2_w,
                                                            wsa_bf, wla_bf, w1_bf, w2_bf);
    k_transpose_in<<<(BB * NN + 255) / 256, 256, 0, stream>>>(pc, pf, coor, fea);
    k_fps<<<BB, 1024, 0, stream>>>(coor, fpsidx);
    k_newcoor<<<(BB * SS + 255) / 256, 256, 0, stream>>>(coor, fpsidx, newc);
    k_knn<<<(BB * SS + 255) / 256, 256, 0, stream>>>(coor, newc, NN, 0.2f * 0.2f, gidx1);
    k_gather_gemm<CC><<<(BB * SS * KK) / 64, 128, 0, stream>>>(
        fea, coor, newc, gidx1, wsa_bf, sa_b, NN, 1.f / 0.2f, nfea);
    k_knn<<<(BB * SS + 255) / 256, 256, 0, stream>>>(newc, newc, SS, 0.4f * 0.4f, gidx2);
    k_gather_gemm<C2><<<(BB * SS * KK) / 64, 128, 0, stream>>>(
        nfea, newc, newc, gidx2, wla_bf, la_b, SS, 1.f / 0.4f, xla);
    dim3 g1((BB * SS) / 64, (C2 * EXPAND) / 128);
    k_gemm<true, false, false><<<g1, 128, 0, stream>>>(xla, w1_bf, pw1_b, nullptr, h, C2, C2 * EXPAND);
    dim3 g2((BB * SS) / 64, 1);
    k_gemm<true, true, true><<<g2, 128, 0, stream>>>(h, w2_bf, pw2_b, nfea,
                                                     out + (size_t)BB * 3 * SS, C2 * EXPAND, C2);
    k_coor_out<<<(BB * SS + 255) / 256, 256, 0, stream>>>(newc, out);
}
